// LinearCRF_21028159881266
// MI455X (gfx1250) — matrix operand order repaired and checked by reading, not compiled, not run on hardware
//
#include <hip/hip_runtime.h>
#include <hip/hip_bf16.h>

#define CRF_B 256
#define CRF_S 512
#define CRF_T 128

#define LOG2E_F 1.4426950408889634f
#define LN2_F   0.6931471805599453f

typedef __attribute__((ext_vector_type(16))) _Float16 v16h;
typedef __attribute__((ext_vector_type(8)))  _Float16 v8h;
typedef __attribute__((ext_vector_type(8)))  float    v8f;

union WmmaOp { v16h v; v8h h[2]; };

// ---------------------------------------------------------------------------
// Kernel 1: E[i][j] = exp(transitions[i][j]) in f16, pre-swizzled into WMMA
// B-operand order: idx = ((w*4 + c)*32 + lane)*16 + h
//   order used here (contiguous k, same as A): K = c*32 + hi*16 + h (hi = lane>=16),
//   N = w*16 + lane%16.
// ---------------------------------------------------------------------------
__global__ __launch_bounds__(256) void crf_prep_expT(const float* __restrict__ trans,
                                                     _Float16* __restrict__ Eswz) {
    int idx = blockIdx.x * 256 + threadIdx.x;        // 0..16383
    int h  = idx & 15;
    int l  = (idx >> 4) & 31;
    int c  = (idx >> 9) & 3;
    int w  = (idx >> 11) & 7;
    int hi = l >> 4;
    int n  = w * 16 + (l & 15);
    int k  = c * 32 + hi * 16 + h;
    Eswz[idx] = (_Float16)__expf(trans[k * CRF_T + n]);
}

// ---------------------------------------------------------------------------
// Kernel 2: gold path score, one thread per batch element.
// Reference indexes transitions[tags[s], tags[s-1]] (cur, prev).
// ---------------------------------------------------------------------------
__global__ __launch_bounds__(256) void crf_gold(const float* __restrict__ emis,
                                                const int* __restrict__ tags,
                                                const int* __restrict__ mask,
                                                const float* __restrict__ trans,
                                                const float* __restrict__ startT,
                                                const float* __restrict__ endT,
                                                float* __restrict__ gold) {
    int b = blockIdx.x * blockDim.x + threadIdx.x;
    if (b >= CRF_B) return;
    const int*   tg = tags + b * CRF_S;
    const int*   mk = mask + b * CRF_S;
    const float* em = emis + (size_t)b * CRF_S * CRF_T;
    int prev = tg[0];
    float sc = startT[prev] + em[prev];
    int cnt = mk[0];
    for (int s = 1; s < CRF_S; ++s) {
        int cur = tg[s];
        float m = (float)mk[s];
        sc += m * (trans[cur * CRF_T + prev] + em[s * CRF_T + cur]);
        cnt += mk[s];
        prev = cur;
    }
    sc += endT[tg[cnt - 1]];
    gold[b] = sc;
}

// ---------------------------------------------------------------------------
// Kernel 3: forward algorithm. One block = 16 batch rows, 8 wave32s = 8
// column tiles. Per step (2 barriers):
//   A: row-max via __shfl_xor (16-lane groups), emitb = emit + M (prefetched),
//      p = exp2((score-M)*log2e) -> f16 LDS
//   C: 4x v_wmma_f32_16x16x32_f16 (B operand resident in VGPRs, 2 chains),
//      score = mask ? emitb + log(acc) : score  (branchless cndmask)
// ---------------------------------------------------------------------------
#define SSTR 132   // score row stride (f32)
#define ESTR 132   // emit row stride (f32)
#define PSTR 136   // p row stride (f16)

__global__ __launch_bounds__(256) void crf_forward(
    const float* __restrict__ emis, const int* __restrict__ mask,
    const float* __restrict__ startT, const float* __restrict__ endT,
    const _Float16* __restrict__ EswzG, float* __restrict__ fwd)
{
    __shared__ __align__(16) float    score[16 * SSTR];   // 8.25 KB
    __shared__ __align__(16) float    emitb[16 * ESTR];   // 8.25 KB
    __shared__ __align__(16) _Float16 pbuf[16 * PSTR];    // 4.25 KB
    __shared__ __align__(16) int      mrow[16];

    const int tid   = threadIdx.x;
    const int lane  = tid & 31;
    const int wv    = tid >> 5;       // wave id -> column tile 0..7
    const int row   = tid >> 4;       // pointwise: batch row 0..15
    const int c16   = tid & 15;
    const int col8  = c16 * 8;
    const int bBase = blockIdx.x * 16;
    const int hi    = lane >> 4;
    const int n16   = lane & 15;

    // Loop-invariant B operand: 4 K-chunks of swizzled exp(transitions),
    // resident in 32 VGPRs for the whole sequence.
    WmmaOp Breg[4];
    #pragma unroll
    for (int c = 0; c < 4; ++c) {
        const _Float16* bp = EswzG + ((wv * 4 + c) * 32 + lane) * 16;
        Breg[c].h[0] = *(const v8h*)bp;
        Breg[c].h[1] = *(const v8h*)(bp + 8);
    }

    // score0 = start_transitions + emissions[:, 0, :]
    const float* erow = emis + (size_t)(bBase + row) * CRF_S * CRF_T + col8;
    {
        #pragma unroll
        for (int u = 0; u < 8; ++u)
            score[row * SSTR + col8 + u] = startT[col8 + u] + erow[u];
    }

    // Prefetch emissions and mask for t = 1.
    float4 e0 = ((const float4*)(erow + CRF_T))[0];
    float4 e1 = ((const float4*)(erow + CRF_T))[1];
    int mCur = 0;
    if (c16 == 0) mCur = mask[(bBase + row) * CRF_S + 1];
    __syncthreads();

    for (int t = 1; t < CRF_S; ++t) {
        // ---- Phase A -------------------------------------------------------
        float sv[8];
        #pragma unroll
        for (int u = 0; u < 8; ++u) sv[u] = score[row * SSTR + col8 + u];
        float m = sv[0];
        #pragma unroll
        for (int u = 1; u < 8; ++u) m = fmaxf(m, sv[u]);
        m = fmaxf(m, __shfl_xor(m, 1, 32));
        m = fmaxf(m, __shfl_xor(m, 2, 32));
        m = fmaxf(m, __shfl_xor(m, 4, 32));
        m = fmaxf(m, __shfl_xor(m, 8, 32));   // all 16 lanes of the row have M

        // emitb = emit + M  (M folded in; Mrow never needed in phase C)
        *((float4*)&emitb[row * ESTR + col8]) =
            make_float4(e0.x + m, e0.y + m, e0.z + m, e0.w + m);
        *((float4*)&emitb[row * ESTR + col8 + 4]) =
            make_float4(e1.x + m, e1.y + m, e1.z + m, e1.w + m);
        if (c16 == 0) mrow[row] = mCur;

        // p = exp2((score - M) * log2e) in f16 (p <= 1)
        v8h ph;
        #pragma unroll
        for (int u = 0; u < 8; ++u)
            ph[u] = (_Float16)__builtin_amdgcn_exp2f((sv[u] - m) * LOG2E_F);
        *((v8h*)&pbuf[row * PSTR + col8]) = ph;

        // Prefetch next step's emissions + mask (hidden behind WMMA phase).
        int tn = (t + 1 < CRF_S) ? (t + 1) : (CRF_S - 1);
        e0 = ((const float4*)(erow + (size_t)tn * CRF_T))[0];
        e1 = ((const float4*)(erow + (size_t)tn * CRF_T))[1];
        if (c16 == 0) mCur = mask[(bBase + row) * CRF_S + tn];
        __syncthreads();

        // ---- Phase C: acc = P(16xK) x E(KxN), K=128, two independent chains
        WmmaOp A[4];
        #pragma unroll
        for (int c = 0; c < 4; ++c) {
            const _Float16* ap = &pbuf[n16 * PSTR + c * 32 + hi * 16];
            A[c].h[0] = *(const v8h*)ap;
            A[c].h[1] = *(const v8h*)(ap + 8);
        }
        v8f acc0 = {}, acc1 = {};
        acc0 = __builtin_amdgcn_wmma_f32_16x16x32_f16(false, A[0].v, false, Breg[0].v,
                                                      (short)0, acc0, false, false);
        acc1 = __builtin_amdgcn_wmma_f32_16x16x32_f16(false, A[2].v, false, Breg[2].v,
                                                      (short)0, acc1, false, false);
        acc0 = __builtin_amdgcn_wmma_f32_16x16x32_f16(false, A[1].v, false, Breg[1].v,
                                                      (short)0, acc0, false, false);
        acc1 = __builtin_amdgcn_wmma_f32_16x16x32_f16(false, A[3].v, false, Breg[3].v,
                                                      (short)0, acc1, false, false);
        v8f acc = acc0 + acc1;

        // score = mask ? (emit + M) + ln(acc) : score   (branchless)
        {
            int j = wv * 16 + n16;
            #pragma unroll
            for (int r = 0; r < 8; ++r) {
                int rr = r + 8 * hi;
                float nxt = emitb[rr * ESTR + j] +
                            __builtin_amdgcn_logf(acc[r]) * LN2_F;
                float old = score[rr * SSTR + j];
                score[rr * SSTR + j] = mrow[rr] ? nxt : old;
            }
        }
        __syncthreads();
    }

    // Final: fwd[b] = logsumexp_j(score[b][j] + endT[j]) via shuffle reduce.
    {
        float v[8];
        float m = -3.0e38f;
        #pragma unroll
        for (int u = 0; u < 8; ++u) {
            v[u] = score[row * SSTR + col8 + u] + endT[col8 + u];
            m = fmaxf(m, v[u]);
        }
        m = fmaxf(m, __shfl_xor(m, 1, 32));
        m = fmaxf(m, __shfl_xor(m, 2, 32));
        m = fmaxf(m, __shfl_xor(m, 4, 32));
        m = fmaxf(m, __shfl_xor(m, 8, 32));
        float s = 0.f;
        #pragma unroll
        for (int u = 0; u < 8; ++u)
            s += __builtin_amdgcn_exp2f((v[u] - m) * LOG2E_F);
        s += __shfl_xor(s, 1, 32);
        s += __shfl_xor(s, 2, 32);
        s += __shfl_xor(s, 4, 32);
        s += __shfl_xor(s, 8, 32);
        if (c16 == 0)
            fwd[bBase + row] = m + __builtin_amdgcn_logf(s) * LN2_F;
    }
}

// ---------------------------------------------------------------------------
// Kernel 4: out = mean(fwd - gold)
// ---------------------------------------------------------------------------
__global__ __launch_bounds__(256) void crf_mean(const float* __restrict__ fwd,
                                                const float* __restrict__ gold,
                                                float* __restrict__ out) {
    __shared__ float red[256];
    int tid = threadIdx.x;
    red[tid] = fwd[tid] - gold[tid];
    __syncthreads();
    for (int s = 128; s > 0; s >>= 1) {
        if (tid < s) red[tid] += red[tid + s];
        __syncthreads();
    }
    if (tid == 0) out[0] = red[0] * (1.0f / CRF_B);
}

extern "C" void kernel_launch(void* const* d_in, const int* in_sizes, int n_in,
                              void* d_out, int out_size, void* d_ws, size_t ws_size,
                              hipStream_t stream) {
    (void)in_sizes; (void)n_in; (void)out_size; (void)ws_size;
    const float* emis   = (const float*)d_in[0];   // (B,S,T) f32
    const int*   tags   = (const int*)d_in[1];     // (B,S) i32
    const int*   mask   = (const int*)d_in[2];     // (B,S) i32
    const float* trans  = (const float*)d_in[3];   // (T,T) f32
    const float* startT = (const float*)d_in[4];   // (T,) f32
    const float* endT   = (const float*)d_in[5];   // (T,) f32
    float* out = (float*)d_out;

    char* ws = (char*)d_ws;
    _Float16* Eswz = (_Float16*)ws;                        // 32768 B
    float*    gold = (float*)(ws + 32768);                 // 1024 B
    float*    fwd  = (float*)(ws + 32768 + 1024);          // 1024 B

    crf_prep_expT<<<64, 256, 0, stream>>>(trans, Eswz);
    crf_gold<<<1, 256, 0, stream>>>(emis, tags, mask, trans, startT, endT, gold);
    crf_forward<<<CRF_B / 16, 256, 0, stream>>>(emis, mask, startT, endT, Eswz, fwd);
    crf_mean<<<1, 256, 0, stream>>>(fwd, gold, out);
}
